// diy_attention_15831249453463
// MI455X (gfx1250) — matrix operand order repaired and checked by reading, not compiled, not run on hardware
//
#include <hip/hip_runtime.h>
#include <cstdint>
#include <cstddef>

// ---------------------------------------------------------------------------
// MI455X (gfx1250) fused attention-last-row pipeline.
//
// Math refactor (exact in real arithmetic):
//   qf  = Wq xq + bq                      (GEMM1, bf16 WMMA)
//   G   = QF @ Wk   (g_b = Wk^T qf_b)     (GEMM2, bf16 WMMA, Wk pre-transposed)
//   s_t = (g.xs + qf.bk)/32  (mask cases via NEG constants + column sums)
//   p   = softmax(s) ; y = sum_s p_s * xs (streaming kernel, X read ONCE)
//   out = Y @ Wv^T + bv                   (GEMM3, bf16 WMMA)
//
// Roofline: ~13 GFLOP of WMMA + 419 MB X stream  ->  HBM-bound (~18 us floor).
// GEMM tiles staged with GLOBAL_LOAD_ASYNC_TO_LDS_B128 (ASYNCcnt-tracked
// LDS DMA) in a double-buffered pipeline overlapped with WMMA compute.
// ---------------------------------------------------------------------------

typedef __bf16 bf16_t;
typedef __bf16 v16bf __attribute__((ext_vector_type(16)));
typedef float  v8f   __attribute__((ext_vector_type(8)));
typedef int    v4i_t __attribute__((ext_vector_type(4)));

#define D_MODEL 1024
#define TIME    50
#define BATCH   2048
#define NEGC    (-100000.0f)

// Async global->LDS DMA path (gfx1250). Guarded so a missing/renamed builtin
// falls back to the proven global_load_b128 + ds_store_b128 path.
#if defined(__AMDGCN__) && __has_builtin(__builtin_amdgcn_global_load_async_to_lds_b128) && \
    __has_builtin(__builtin_amdgcn_s_wait_asynccnt)
#define USE_ASYNC_LDS 1
#else
#define USE_ASYNC_LDS 0
#endif

// Builtin signature: (v4i AS1* src, v4i AS3* dst, imm offset, imm cpol)
typedef __attribute__((address_space(1))) v4i_t as1_v4i;
typedef __attribute__((address_space(3))) v4i_t as3_v4i;

// Copy 16 bytes (8 bf16) global -> LDS.
__device__ __forceinline__ void stage16(const bf16_t* __restrict__ g, bf16_t* l) {
#if USE_ASYNC_LDS
  __builtin_amdgcn_global_load_async_to_lds_b128((as1_v4i*)g, (as3_v4i*)l, 0, 0);
#else
  *(uint4*)l = *(const uint4*)g;
#endif
}

__device__ __forceinline__ void stage_fence() {
#if USE_ASYNC_LDS
  __builtin_amdgcn_s_wait_asynccnt(0);
#endif
}

// ---- workspace layout (bytes) ----
static constexpr size_t OFF_WQ    = 0;                      // bf16 [1024,1024]  2MB
static constexpr size_t OFF_WKT   = 2ull  << 20;            // bf16 [1024,1024]  2MB (Wk^T)
static constexpr size_t OFF_WV    = 4ull  << 20;            // bf16 [1024,1024]  2MB
static constexpr size_t OFF_XL    = 6ull  << 20;            // bf16 [2048,1024]  4MB (x_last+pe)
static constexpr size_t OFF_QF    = 10ull << 20;            // bf16 [2048,1024]  4MB
static constexpr size_t OFF_G     = 14ull << 20;            // f32  [2048,1024]  8MB
static constexpr size_t OFF_Y     = 22ull << 20;            // bf16 [2048,1024]  4MB
static constexpr size_t OFF_WSK   = 26ull << 20;            // f32  [1024]       4KB (col sums of Wk)
static constexpr size_t OFF_SCAL  = OFF_WSK + 4096;         // f32  [8] (bk_sum)

// ---------------------------------------------------------------------------
// Prep: bf16 conversions of Wq, Wv, and x_last(+pe[49]).
// ---------------------------------------------------------------------------
__global__ void prep_cvt(const float* __restrict__ wq, const float* __restrict__ wv,
                         const float* __restrict__ x,  const float* __restrict__ pe,
                         bf16_t* __restrict__ wqB, bf16_t* __restrict__ wvB,
                         bf16_t* __restrict__ xlB) {
  int i = blockIdx.x * blockDim.x + threadIdx.x;           // 0 .. 4M-1
  if (i < 1048576) {
    wqB[i] = (bf16_t)wq[i];
  } else if (i < 2097152) {
    int j = i - 1048576;
    wvB[j] = (bf16_t)wv[j];
  } else {
    int j = i - 2097152;                                   // 0 .. 2M-1
    int b = j >> 10, d = j & 1023;
    xlB[j] = (bf16_t)(x[(size_t)(b * TIME + (TIME - 1)) * D_MODEL + d] +
                      pe[(TIME - 1) * D_MODEL + d]);
  }
}

// ---------------------------------------------------------------------------
// Transpose Wk (f32 row-major [e][d]) -> WkT bf16 [d][e] so GEMM2 is NT-clean.
// ---------------------------------------------------------------------------
__global__ void transpose_to_bf16(const float* __restrict__ src, bf16_t* __restrict__ dst) {
  __shared__ float tl[32][33];
  const int bx = blockIdx.x * 32;   // source row base (e)
  const int by = blockIdx.y * 32;   // source col base (d)
  const int lx = threadIdx.x & 31;
  const int ly = threadIdx.x >> 5;  // 0..7
#pragma unroll
  for (int p = 0; p < 4; ++p) {
    int r = ly + p * 8;
    tl[r][lx] = src[(size_t)(bx + r) * D_MODEL + by + lx];
  }
  __syncthreads();
#pragma unroll
  for (int p = 0; p < 4; ++p) {
    int r = ly + p * 8;
    dst[(size_t)(by + r) * D_MODEL + bx + lx] = (bf16_t)tl[lx][r];
  }
}

// ---------------------------------------------------------------------------
// Column sums of Wk (for masked-q scores) and sum(bk).
// ---------------------------------------------------------------------------
__global__ void prep_misc(const float* __restrict__ wk, const float* __restrict__ bk,
                          float* __restrict__ wsk, float* __restrict__ scal) {
  if (blockIdx.x < 4) {
    int d = blockIdx.x * 256 + threadIdx.x;
    float s = 0.f;
    for (int e = 0; e < D_MODEL; ++e) s += wk[(size_t)e * D_MODEL + d];
    wsk[d] = s;
  } else {
    __shared__ float r[256];
    float p = 0.f;
    for (int e = threadIdx.x; e < D_MODEL; e += 256) p += bk[e];
    r[threadIdx.x] = p;
    __syncthreads();
    for (int o = 128; o; o >>= 1) {
      if (threadIdx.x < o) r[threadIdx.x] += r[threadIdx.x + o];
      __syncthreads();
    }
    if (threadIdx.x == 0) scal[0] = r[0];
  }
}

// ---------------------------------------------------------------------------
// Unified NT GEMM:  C[m,n] = sum_k A[m,k]*B[n,k] (+bias[n])
// A: bf16 [M,K] row-major, B: bf16 [N,K] row-major (torch-Linear layout).
// 128x128 block, BK=32, 256 threads = 8 waves, each wave 64x32 (8 WMMA accs).
// Double-buffered LDS tiles filled by async global->LDS DMA (ASYNCcnt),
// overlapped with the WMMA compute of the previous K-step.
// Fragment loads are contiguous 16B LDS chunks matching the CDNA5 layouts:
//   A 16x32 bf16: lane(m,h): elems = A[m][h*8+0..7] ++ A[m][16+h*8+0..7]
//   B 32x16 bf16: lane(n,h): elems = B[n][h*8+0..7] ++ B[n][16+h*8+0..7]
// ---------------------------------------------------------------------------
__launch_bounds__(256)
__global__ void gemm_nt_bf16(const bf16_t* __restrict__ A, const bf16_t* __restrict__ B,
                             const float* __restrict__ bias,
                             float* __restrict__ Cf, bf16_t* __restrict__ Cb,
                             int M, int N, int K) {
  constexpr int LDA  = 40;                      // 32 + 8 bf16 pad (80B row, 16B aligned)
  constexpr int TSZ  = 128 * LDA;               // one tile buffer (bf16 elems)
  __shared__ bf16_t As[2 * TSZ];
  __shared__ bf16_t Bs[2 * TSZ];

  const int tid  = threadIdx.x;
  const int lane = tid & 31;
  const int wid  = tid >> 5;
  const int wm   = wid & 1;                     // 2 M-slabs of 64
  const int wn   = wid >> 1;                    // 4 N-slabs of 32
  const int m0   = blockIdx.y * 128;
  const int n0   = blockIdx.x * 128;

  v8f acc[4][2];
#pragma unroll
  for (int i = 0; i < 4; ++i)
#pragma unroll
    for (int j = 0; j < 2; ++j)
#pragma unroll
      for (int e = 0; e < 8; ++e) acc[i][j][e] = 0.f;

  const int lr = tid >> 1;                      // 0..127 tile row
  const int lh = (tid & 1) * 16;                // k half (elements)

  const int mrow = lane & 15;
  const int half = lane >> 4;

  // Stage K-step kt into buffer `buf` (each thread DMAs 2x16B per matrix).
  auto stage = [&](int kt, int buf) {
    const bf16_t* ga = A + (size_t)(m0 + lr) * K + kt * 32 + lh;
    const bf16_t* gb = B + (size_t)(n0 + lr) * K + kt * 32 + lh;
    bf16_t* la = As + buf * TSZ + lr * LDA + lh;
    bf16_t* lb = Bs + buf * TSZ + lr * LDA + lh;
    stage16(ga,     la);
    stage16(ga + 8, la + 8);
    stage16(gb,     lb);
    stage16(gb + 8, lb + 8);
  };

  const int nk = K / 32;
  stage(0, 0);                                  // prologue

  for (int kt = 0; kt < nk; ++kt) {
    const int cur = kt & 1;
    stage_fence();                              // this wave's DMA for tile kt done
    __syncthreads();                            // -> all waves' tiles visible,
                                                //    all reads of buf cur^1 retired
    if (kt + 1 < nk) stage(kt + 1, cur ^ 1);    // overlap next DMA with compute

    union Frag { v16bf v; uint4 q[2]; };
    Frag a[4], bb[2];
    const bf16_t* Ab = As + cur * TSZ;
    const bf16_t* Bb = Bs + cur * TSZ;
#pragma unroll
    for (int i = 0; i < 4; ++i) {
      const bf16_t* p = Ab + (wm * 64 + i * 16 + mrow) * LDA + half * 8;
      a[i].q[0] = *(const uint4*)p;             // K = h*8 .. h*8+7
      a[i].q[1] = *(const uint4*)(p + 16);      // K = 16+h*8 .. 16+h*8+7
    }
#pragma unroll
    for (int j = 0; j < 2; ++j) {
      const bf16_t* p = Bb + (wn * 32 + j * 16 + mrow) * LDA + half * 8;
      bb[j].q[0] = *(const uint4*)p;            // K = h*8 .. h*8+7
      bb[j].q[1] = *(const uint4*)(p + 16);     // K = 16+h*8 .. 16+h*8+7
    }
#pragma unroll
    for (int i = 0; i < 4; ++i)
#pragma unroll
      for (int j = 0; j < 2; ++j)
        acc[i][j] = __builtin_amdgcn_wmma_f32_16x16x32_bf16(
            false, a[i].v, false, bb[j].v, (short)0, acc[i][j], false, false);
  }

#pragma unroll
  for (int i = 0; i < 4; ++i) {
#pragma unroll
    for (int j = 0; j < 2; ++j) {
      const int n = n0 + wn * 32 + j * 16 + (lane & 15);
      const float bv = bias ? bias[n] : 0.f;
#pragma unroll
      for (int vg = 0; vg < 8; ++vg) {
        const int m = m0 + wm * 64 + i * 16 + (lane >> 4) * 8 + vg;
        const float v = acc[i][j][vg] + bv;
        if (Cf) Cf[(size_t)m * N + n] = v;
        else    Cb[(size_t)m * N + n] = (bf16_t)v;
      }
    }
  }
}

// ---------------------------------------------------------------------------
// Streaming attention: one WG per batch. Stages x_b+pe (200KB f32) in LDS,
// computes scores (mask cases), softmax, and y = sum_s p_s*xs -> bf16 Y.
// This is the single pass over X (419MB) = the HBM roofline of the problem.
// (+pe fusion keeps this on the VGPR path; it is bandwidth-bound either way.)
// ---------------------------------------------------------------------------
__launch_bounds__(512)
__global__ void attention_fused(const float* __restrict__ x, const int* __restrict__ mask,
                                const float* __restrict__ bk, const float* __restrict__ pe,
                                const float* __restrict__ G, const bf16_t* __restrict__ QFB,
                                const float* __restrict__ wsk, const float* __restrict__ scal,
                                bf16_t* __restrict__ YB) {
  extern __shared__ char smem[];
  float* Xs  = (float*)smem;               // [50*1024]
  float* Gs  = Xs  + TIME * D_MODEL;       // [1024]
  float* QFs = Gs  + D_MODEL;              // [1024]
  float* WKs = QFs + D_MODEL;              // [1024]
  float* Sc  = WKs + D_MODEL;              // [64]  S1 / final score
  float* S2  = Sc  + 64;                   // [64]  wsum_k . xs
  float* Pp  = S2  + 64;                   // [64]  softmax probs
  float* Rr  = Pp  + 64;                   // [8]   {t3=qf.bk, s4=sum(qf)}

  const int b    = blockIdx.x;
  const int t    = threadIdx.x;
  const int lane = t & 31;
  const int wid  = t >> 5;

  // ---- stage x_b + pe, g_b, qf_b, wsum_k ----
  const float4* x4  = (const float4*)(x + (size_t)b * TIME * D_MODEL);
  const float4* pe4 = (const float4*)pe;
  float4* Xs4 = (float4*)Xs;
#pragma unroll
  for (int i = 0; i < 25; ++i) {
    int idx = t + i * 512;                 // 12800 float4
    float4 a = x4[idx], p = pe4[idx];
    a.x += p.x; a.y += p.y; a.z += p.z; a.w += p.w;
    Xs4[idx] = a;
  }
  if (t < 256) {
    ((float4*)Gs)[t]  = ((const float4*)(G + (size_t)b * D_MODEL))[t];
    ((float4*)WKs)[t] = ((const float4*)wsk)[t];
  }
  {
    int d = t * 2;
    QFs[d]     = (float)QFB[(size_t)b * D_MODEL + d];
    QFs[d + 1] = (float)QFB[(size_t)b * D_MODEL + d + 1];
  }
  __syncthreads();

  // ---- dot products: S1[s]=g.xs, S2[s]=wsum_k.xs (one wave per row) ----
  const float4* Xc = (const float4*)Xs;
  const float4* Gc = (const float4*)Gs;
  const float4* Wc = (const float4*)WKs;
  for (int s = wid; s < TIME; s += 16) {
    float p1 = 0.f, p2 = 0.f;
#pragma unroll
    for (int c = 0; c < 8; ++c) {
      float4 xv = Xc[s * 256 + c * 32 + lane];
      float4 gv = Gc[c * 32 + lane];
      float4 wv = Wc[c * 32 + lane];
      p1 += xv.x * gv.x + xv.y * gv.y + xv.z * gv.z + xv.w * gv.w;
      p2 += xv.x * wv.x + xv.y * wv.y + xv.z * wv.z + xv.w * wv.w;
    }
    for (int o = 16; o; o >>= 1) { p1 += __shfl_xor(p1, o, 32); p2 += __shfl_xor(p2, o, 32); }
    if (lane == 0) { Sc[s] = p1; S2[s] = p2; }
  }
  if (wid == 0) {                          // t3 = qf . bk
    float p = 0.f;
    for (int d = lane; d < D_MODEL; d += 32) p += QFs[d] * bk[d];
    for (int o = 16; o; o >>= 1) p += __shfl_xor(p, o, 32);
    if (lane == 0) Rr[0] = p;
  } else if (wid == 1) {                   // s4 = sum(qf)
    float p = 0.f;
    for (int d = lane; d < D_MODEL; d += 32) p += QFs[d];
    for (int o = 16; o; o >>= 1) p += __shfl_xor(p, o, 32);
    if (lane == 0) Rr[1] = p;
  }
  __syncthreads();

  // ---- mask cases + scale ----
  if (t < TIME) {
    const int mq = mask[b * TIME + (TIME - 1)];
    const int ms = mask[b * TIME + t];
    float sc;
    if (mq && ms)  sc = (Sc[t] + Rr[0]) * 0.03125f;               // qf.kf / 32
    else if (mq)   sc = (NEGC * Rr[1]) * 0.03125f;                // k masked
    else if (ms)   sc = (NEGC * (S2[t] + scal[0])) * 0.03125f;    // q masked
    else           sc = (NEGC * NEGC * (float)D_MODEL) * 0.03125f;
    Sc[t] = sc;
  }
  __syncthreads();

  // ---- softmax over 50 (wave 0) ----
  if (wid == 0) {
    float v0 = Sc[lane];
    float v1 = (lane + 32 < TIME) ? Sc[lane + 32] : -3.4e38f;
    float mx = fmaxf(v0, v1);
    for (int o = 16; o; o >>= 1) mx = fmaxf(mx, __shfl_xor(mx, o, 32));
    float e0 = __expf(v0 - mx);
    float e1 = (lane + 32 < TIME) ? __expf(v1 - mx) : 0.f;
    float sm = e0 + e1;
    for (int o = 16; o; o >>= 1) sm += __shfl_xor(sm, o, 32);
    float inv = 1.f / sm;
    Pp[lane] = e0 * inv;
    if (lane + 32 < TIME) Pp[lane + 32] = e1 * inv;
  }
  __syncthreads();

  // ---- y = sum_s p_s * xs  ->  bf16 ----
#pragma unroll
  for (int i = 0; i < 2; ++i) {
    int d = t + i * 512;
    float acc = 0.f;
#pragma unroll
    for (int s = 0; s < TIME; ++s) acc += Pp[s] * Xs[s * D_MODEL + d];
    YB[(size_t)b * D_MODEL + d] = (bf16_t)acc;
  }
}

// ---------------------------------------------------------------------------
extern "C" void kernel_launch(void* const* d_in, const int* in_sizes, int n_in,
                              void* d_out, int out_size, void* d_ws, size_t ws_size,
                              hipStream_t stream) {
  const float* x    = (const float*)d_in[0];
  const int*   mask = (const int*)  d_in[1];
  const float* wq   = (const float*)d_in[2];
  const float* bq   = (const float*)d_in[3];
  const float* wk   = (const float*)d_in[4];
  const float* bk   = (const float*)d_in[5];
  const float* wv   = (const float*)d_in[6];
  const float* bv   = (const float*)d_in[7];
  const float* pe   = (const float*)d_in[8];

  char* ws = (char*)d_ws;
  bf16_t* WqB  = (bf16_t*)(ws + OFF_WQ);
  bf16_t* WkT  = (bf16_t*)(ws + OFF_WKT);
  bf16_t* WvB  = (bf16_t*)(ws + OFF_WV);
  bf16_t* XlB  = (bf16_t*)(ws + OFF_XL);
  bf16_t* QFB  = (bf16_t*)(ws + OFF_QF);
  float*  Gf   = (float*) (ws + OFF_G);
  bf16_t* YB   = (bf16_t*)(ws + OFF_Y);
  float*  WSK  = (float*) (ws + OFF_WSK);
  float*  SCAL = (float*) (ws + OFF_SCAL);

  // prep: bf16 weights, x_last(+pe), Wk transpose, column sums
  prep_cvt<<<16384, 256, 0, stream>>>(wq, wv, x, pe, WqB, WvB, XlB);
  transpose_to_bf16<<<dim3(32, 32), 256, 0, stream>>>(wk, WkT);
  prep_misc<<<5, 256, 0, stream>>>(wk, bk, WSK, SCAL);

  // GEMM1: QF = Xlast @ Wq^T + bq   (bf16 out)
  gemm_nt_bf16<<<dim3(8, 16), 256, 0, stream>>>(XlB, WqB, bq, nullptr, QFB,
                                                BATCH, D_MODEL, D_MODEL);
  // GEMM2: G = QF @ Wk  (via WkT, NT-clean, f32 out)
  gemm_nt_bf16<<<dim3(8, 16), 256, 0, stream>>>(QFB, WkT, nullptr, Gf, nullptr,
                                                BATCH, D_MODEL, D_MODEL);

  // fused scores/softmax/y: one pass over X
  size_t smem = (size_t)(TIME * D_MODEL + 3 * D_MODEL + 3 * 64 + 8) * sizeof(float);
  attention_fused<<<BATCH, 512, smem, stream>>>(x, mask, bk, pe, Gf, QFB, WSK, SCAL, YB);

  // GEMM3: out = Y @ Wv^T + bv  (f32 -> d_out)
  gemm_nt_bf16<<<dim3(8, 16), 256, 0, stream>>>(YB, WvB, bv, (float*)d_out, nullptr,
                                                BATCH, D_MODEL, D_MODEL);
}
